// DeepSeekV3MoEGate_77773267796129
// MI455X (gfx1250) — matrix operand order repaired and checked by reading, not compiled, not run on hardware
//
#include <hip/hip_runtime.h>

// ---------------------------------------------------------------------------
// DeepSeek-V3 MoE gate for MI455X (gfx1250, wave32, WMMA + TDM)
//   x:[2,4096,7168] f32, gate_weight:[256,7168] f32, bias:[256] f32
//   out: weights[8192*8] f32 | indices[8192*8] (as float) | aux_loss[1]
//
// GEMM path: split-bf16 (hi/lo truncation split, 3 WMMA products ~= fp32)
// W staged per 32-k chunk (32KB: 8192 hi-bf16 ++ 8192 lo-bf16) into LDS by the
// Tensor Data Mover, double buffered, one s_barrier per k-step.
// ---------------------------------------------------------------------------

typedef __bf16 v16bf __attribute__((ext_vector_type(16)));
typedef float  v8f   __attribute__((ext_vector_type(8)));
typedef unsigned int u32x4 __attribute__((ext_vector_type(4)));
typedef int i32x8 __attribute__((ext_vector_type(8)));
typedef int i32x4 __attribute__((ext_vector_type(4)));

#define DIMX  7168
#define NEXP  256
#define NKC   224          // DIMX / 32 k-chunks
#define NTOK  8192
#define CHUNK_ELEMS 16384  // bf16 per chunk: 8192 hi + 8192 lo (32 KB)

#if defined(__has_builtin)
#if __has_builtin(__builtin_amdgcn_tensor_load_to_lds) && \
    __has_builtin(__builtin_amdgcn_s_wait_tensorcnt)
#define USE_TDM 1
#endif
#endif
#ifndef USE_TDM
#define USE_TDM 0
#endif

union V16 {
  v16bf v;
  unsigned short u[16];
  uint4 q[2];
};

// ---------------------------------------------------------------------------
// Kernel 1: split gate_weight into hi/lo bf16 with exact truncation split,
// permuted into B-fragment-major chunks:
//   chunk kc (32 k-values): [ (ntile*2+half)*16 + n_local ][16 k]  hi then lo
// B(32x16) lane L is fed W[n = L%16][k = (L/16)*16 .. +16) (contiguous k, same order as A).
// ---------------------------------------------------------------------------
__global__ void __launch_bounds__(256)
split_w_kernel(const float* __restrict__ W,
               unsigned short* __restrict__ wsplit) {
  int idx = blockIdx.x * blockDim.x + threadIdx.x;   // 256*224*2 = 114688
  int h  = idx & 1;
  int kc = (idx >> 1) % NKC;
  int e  = idx / (2 * NKC);
  if (e >= NEXP) return;
  int t  = e >> 4;
  int nl = e & 15;
  const float* src = W + (size_t)e * DIMX + kc * 32 + h * 16;
  size_t dhi = (size_t)kc * CHUNK_ELEMS + (size_t)(((t * 2 + h) * 16 + nl)) * 16;
  size_t dlo = dhi + 8192;
  #pragma unroll
  for (int j = 0; j < 16; ++j) {
    float xv = src[j];
    unsigned int b = __float_as_uint(xv);
    float hif = __uint_as_float(b & 0xFFFF0000u);   // exact truncation split
    float lof = xv - hif;                           // exact in f32
    wsplit[dhi + j] = (unsigned short)(b >> 16);
    wsplit[dlo + j] = (unsigned short)(__float_as_uint(lof) >> 16);
  }
}

// ---------------------------------------------------------------------------
// TDM: one descriptor moves a whole 32KB W chunk into an LDS buffer.
// D# group0: count=1 | lds_addr | global_addr | type=2
// D# group1: data_size=8B, tensor_dim0=4096, tensor_dim1=1, tile_dim0=4096,
//            tile_dim1=1, dim0_stride=4096; remaining groups zero (<=2D).
// ---------------------------------------------------------------------------
#if USE_TDM
__device__ __forceinline__ void tdm_load_chunk(unsigned ldsOff,
                                               const unsigned short* wsplit,
                                               int kc) {
  unsigned long long ga = (unsigned long long)(size_t)wsplit +
                          (unsigned long long)kc * 32768ull;
  u32x4 g0;
  g0[0] = 1u;                                             // count=1, user D#
  g0[1] = ldsOff;                                         // lds_addr (bytes)
  g0[2] = (unsigned)(ga & 0xffffffffull);                 // global_addr[31:0]
  g0[3] = (unsigned)((ga >> 32) & 0x01ffffffull)          // global_addr[56:32]
          | (2u << 30);                                   // type=2 ("image")
  i32x8 g1;
  g1[0] = (int)(3u << 16);      // wg_mask=0 (no cluster), data_size=3 (8B)
  g1[1] = (int)(4096u << 16);   // tensor_dim0 = 4096 (x8B = 32KB)
  g1[2] = (int)(1u << 16);      // tensor_dim0 hi=0 | tensor_dim1 = 1
  g1[3] = (int)(4096u << 16);   // tensor_dim1 hi=0 | tile_dim0 = 4096
  g1[4] = 1;                    // tile_dim1 = 1, tile_dim2 = 0
  g1[5] = 4096;                 // tensor_dim0_stride lo32
  g1[6] = 0;
  g1[7] = 0;
  i32x4 z4;
  z4[0] = 0; z4[1] = 0; z4[2] = 0; z4[3] = 0;
  i32x8 z8;
  z8[0] = 0; z8[1] = 0; z8[2] = 0; z8[3] = 0;
  z8[4] = 0; z8[5] = 0; z8[6] = 0; z8[7] = 0;
  // 6-arg form (clang-23 / therock HIP headers): groups 0..4 + cpol
  __builtin_amdgcn_tensor_load_to_lds(g0, g1, z4, z4, z8, 0);
}
#endif

// ---------------------------------------------------------------------------
// Kernel 2: logits GEMM with split-bf16 WMMA.
// Block: 256 thr = 8 waves; tile 64 tokens x 256 experts.
// Wave: 2 M-tiles(16) x 4 N-tiles(16).  K staged in LDS, double buffered.
// ---------------------------------------------------------------------------
__global__ void __launch_bounds__(256)
gate_gemm_kernel(const float* __restrict__ x,
                 const unsigned short* __restrict__ wsplit,
                 float* __restrict__ logits) {
  extern __shared__ unsigned short smem[];   // 2 x 16384 bf16 = 64 KB

  const int tid  = threadIdx.x;
  const int wave = tid >> 5;
  const int lane = tid & 31;
  const int mp   = wave & 1;    // which pair of M-tiles (0..1)
  const int nq   = wave >> 1;   // which quarter of N-tiles (0..3)
  const int ml   = lane & 15;
  const int half = lane >> 4;

  const size_t row0 = (size_t)(blockIdx.x * 64 + mp * 32 + ml);
  const float* xr0 = x + row0 * DIMX;
  const float* xr1 = xr0 + (size_t)16 * DIMX;

  v8f acc[2][4];
  #pragma unroll
  for (int m = 0; m < 2; ++m)
    #pragma unroll
    for (int t = 0; t < 4; ++t)
      #pragma unroll
      for (int r = 0; r < 8; ++r)
        acc[m][t][r] = 0.0f;

#if USE_TDM
  const unsigned ldsBase = (unsigned)(size_t)(void*)smem;
  // prologue: DMA chunk 0 into buffer 0 (one TDM op, wave 0 only)
  if (wave == 0) tdm_load_chunk(ldsBase, wsplit, 0);
#else
  // fallback prologue: cooperative copy of chunk 0 into buffer 0
  {
    const uint4* g = (const uint4*)(wsplit);
    #pragma unroll
    for (int j = 0; j < 8; ++j)
      ((uint4*)smem)[tid * 8 + j] = g[tid * 8 + j];
  }
#endif

  for (int kc = 0; kc < NKC; ++kc) {
#if USE_TDM
    // chunk kc landed? (TENSORcnt of non-issuing waves is trivially 0)
    __builtin_amdgcn_s_wait_tensorcnt(0);
#endif
    // One barrier: publishes chunk kc AND proves buf[(kc+1)&1] is free.
    __syncthreads();
#if USE_TDM
    if (wave == 0 && kc + 1 < NKC)
      tdm_load_chunk(ldsBase + (unsigned)(((kc + 1) & 1) * 32768),
                     wsplit, kc + 1);
#else
    if (kc + 1 < NKC) {
      const uint4* g = (const uint4*)(wsplit + (size_t)(kc + 1) * CHUNK_ELEMS);
      uint4* sdst = (uint4*)(smem + ((kc + 1) & 1) * CHUNK_ELEMS);
      #pragma unroll
      for (int j = 0; j < 8; ++j)
        sdst[tid * 8 + j] = g[tid * 8 + j];
    }
#endif
    const unsigned short* whi_s = smem + (kc & 1) * CHUNK_ELEMS;
    const unsigned short* wlo_s = whi_s + 8192;

    // ---- A fragments: contiguous k order (same as B): lane(m=L%16, half=L/16)
    //      elems 0..15 -> k = kc*32 + half*16 + j
    V16 ah[2], al[2];
    #pragma unroll
    for (int m = 0; m < 2; ++m) {
      const float* ap = (m ? xr1 : xr0) + kc * 32 + half * 16;
      float4 f0 = *(const float4*)(ap);
      float4 f1 = *(const float4*)(ap + 4);
      float4 f2 = *(const float4*)(ap + 8);
      float4 f3 = *(const float4*)(ap + 12);
      float fv[16] = {f0.x, f0.y, f0.z, f0.w, f1.x, f1.y, f1.z, f1.w,
                      f2.x, f2.y, f2.z, f2.w, f3.x, f3.y, f3.z, f3.w};
      #pragma unroll
      for (int j = 0; j < 16; ++j) {
        unsigned int b = __float_as_uint(fv[j]);
        ah[m].u[j] = (unsigned short)(b >> 16);
        float lof = fv[j] - __uint_as_float(b & 0xFFFF0000u);
        al[m].u[j] = (unsigned short)(__float_as_uint(lof) >> 16);
      }
    }

    // ---- 4 N-tiles x 2 M-tiles x 3 split products = 24 WMMAs / k-step
    #pragma unroll
    for (int t4 = 0; t4 < 4; ++t4) {
      const int t = nq * 4 + t4;
      const int u = ((t * 2 + half) * 16 + ml) * 16;   // bf16 elem offset
      V16 bh, bl;
      bh.q[0] = *(const uint4*)(whi_s + u);
      bh.q[1] = *(const uint4*)(whi_s + u + 8);
      bl.q[0] = *(const uint4*)(wlo_s + u);
      bl.q[1] = *(const uint4*)(wlo_s + u + 8);
      #pragma unroll
      for (int m = 0; m < 2; ++m) {
        acc[m][t4] = __builtin_amdgcn_wmma_f32_16x16x32_bf16(
            false, ah[m].v, false, bh.v, (short)0, acc[m][t4], false, false);
        acc[m][t4] = __builtin_amdgcn_wmma_f32_16x16x32_bf16(
            false, al[m].v, false, bh.v, (short)0, acc[m][t4], false, false);
        acc[m][t4] = __builtin_amdgcn_wmma_f32_16x16x32_bf16(
            false, ah[m].v, false, bl.v, (short)0, acc[m][t4], false, false);
      }
    }
  }

  // ---- store C: VGPR r -> M = r + 8*half, N = lane%16
  #pragma unroll
  for (int m = 0; m < 2; ++m) {
    const int mbase = blockIdx.x * 64 + mp * 32 + m * 16 + half * 8;
    #pragma unroll
    for (int t4 = 0; t4 < 4; ++t4) {
      const int n = (nq * 4 + t4) * 16 + ml;
      #pragma unroll
      for (int r = 0; r < 8; ++r)
        logits[(size_t)(mbase + r) * NEXP + n] = acc[m][t4][r];
    }
  }
}

// ---------------------------------------------------------------------------
// Kernel 3: per-token sigmoid + grouped top-2/top-4 + top-8 + aux-loss accum.
// One wave per token (8 tokens per wave via loop); 64 tokens per block.
// ---------------------------------------------------------------------------
__global__ void __launch_bounds__(256)
topk_kernel(const float* __restrict__ logits,
            const float* __restrict__ bias,
            float* __restrict__ wout,    // [NTOK*8]
            float* __restrict__ iout,    // [NTOK*8] expert ids as float
            float* __restrict__ gsum,    // [2*NEXP] sum of sigmoid scores
            float* __restrict__ gcnt) {  // [2*NEXP] selection counts
  __shared__ float sum_s[NEXP];
  __shared__ float cnt_s[NEXP];
  const int tid  = threadIdx.x;
  const int wave = tid >> 5;
  const int lane = tid & 31;
  sum_s[tid] = 0.0f;
  cnt_s[tid] = 0.0f;
  __syncthreads();

  float4 b0 = *(const float4*)(bias + lane * 8);
  float4 b1 = *(const float4*)(bias + lane * 8 + 4);
  const float bj[8] = {b0.x, b0.y, b0.z, b0.w, b1.x, b1.y, b1.z, b1.w};
  const int g = lane >> 2;           // expert-group of this lane's 8 experts
  float ssum[8] = {0, 0, 0, 0, 0, 0, 0, 0};

  for (int it = 0; it < 8; ++it) {
    const int token = blockIdx.x * 64 + wave * 8 + it;
    const float* lr = logits + (size_t)token * NEXP + lane * 8;
    float4 z0 = *(const float4*)lr;
    float4 z1 = *(const float4*)(lr + 4);
    const float zz[8] = {z0.x, z0.y, z0.z, z0.w, z1.x, z1.y, z1.z, z1.w};
    float s[8], sb[8];
    #pragma unroll
    for (int j = 0; j < 8; ++j) {
      s[j] = 1.0f / (1.0f + __expf(-zz[j]));   // original sigmoid score
      ssum[j] += s[j];
      sb[j] = s[j] + bj[j];                    // biased score for selection
    }

    // per-lane top-2 of 8 biased scores
    float m1 = -1e30f, m2 = -1e30f;
    #pragma unroll
    for (int j = 0; j < 8; ++j) {
      float v = sb[j];
      float n1 = fmaxf(m1, v);
      m2 = fmaxf(m2, fminf(m1, v));
      m1 = n1;
    }
    // merge top-2 across the 4 lanes of the group (quad shuffle)
    #pragma unroll
    for (int off = 1; off <= 2; off <<= 1) {
      float o1 = __shfl_xor(m1, off);
      float o2 = __shfl_xor(m2, off);
      float n1 = fmaxf(m1, o1);
      float n2 = fmaxf(fminf(m1, o1), fmaxf(m2, o2));
      m1 = n1; m2 = n2;
    }
    const float gsc = m1 + m2;  // group score (top-2 sum)

    // rank my group among 8; keep top-4 (tie -> lower group index)
    int rank = 0;
    #pragma unroll
    for (int g2 = 0; g2 < 8; ++g2) {
      float og = __shfl(gsc, g2 * 4);
      rank += (og > gsc || (og == gsc && g2 < g)) ? 1 : 0;
    }
    const bool sel = (rank < 4);
    float msk[8];
    #pragma unroll
    for (int j = 0; j < 8; ++j) msk[j] = sel ? sb[j] : -1e30f;

    // top-8 experts: 8 rounds of full-wave butterfly argmax
    unsigned rem = 0;
    float wsum = 0.0f;
    float wv[8]; int wi[8];
    #pragma unroll
    for (int r = 0; r < 8; ++r) {
      float bv = -2e30f, bo = 0.0f; int bjx = 0;
      #pragma unroll
      for (int j = 0; j < 8; ++j) {
        bool ok = (((rem >> j) & 1u) == 0u) && (msk[j] > bv);
        bv  = ok ? msk[j] : bv;
        bo  = ok ? s[j]   : bo;
        bjx = ok ? j      : bjx;
      }
      int bidx = lane * 8 + bjx;
      #pragma unroll
      for (int off = 16; off >= 1; off >>= 1) {
        float ov = __shfl_xor(bv, off);
        float oo = __shfl_xor(bo, off);
        int   oi = __shfl_xor(bidx, off);
        bool take = (ov > bv) || (ov == bv && oi < bidx);  // tie -> low index
        bv = take ? ov : bv;
        bo = take ? oo : bo;
        bidx = take ? oi : bidx;
      }
      wv[r] = bo; wi[r] = bidx; wsum += bo;
      if ((bidx >> 3) == lane) rem |= (1u << (bidx & 7));
    }

    if (lane == 0) {
      const float sc = 2.5f / wsum;   // normalize + route_scale
      #pragma unroll
      for (int r = 0; r < 8; ++r) {
        wout[token * 8 + r] = wv[r] * sc;
        iout[token * 8 + r] = (float)wi[r];
        atomicAdd(&cnt_s[wi[r]], 1.0f);
      }
    }
  }

  // fold per-lane score sums into LDS, then one global add per expert
  #pragma unroll
  for (int j = 0; j < 8; ++j) atomicAdd(&sum_s[lane * 8 + j], ssum[j]);
  __syncthreads();
  const int b = blockIdx.x >> 6;   // 64 blocks per batch of 4096 tokens
  atomicAdd(&gsum[b * NEXP + tid], sum_s[tid]);
  atomicAdd(&gcnt[b * NEXP + tid], cnt_s[tid]);
}

// ---------------------------------------------------------------------------
// Kernel 4: aux = alpha * mean_b( sum_e (cnt/128) * (sum/4096) )
// ---------------------------------------------------------------------------
__global__ void __launch_bounds__(256)
finalize_kernel(const float* __restrict__ gsum,
                const float* __restrict__ gcnt,
                float* __restrict__ aux_out) {
  __shared__ float red[256];
  const int t = threadIdx.x;
  float p = 0.0f;
  #pragma unroll
  for (int b = 0; b < 2; ++b)
    p += (gcnt[b * NEXP + t] * (1.0f / 128.0f)) *
         (gsum[b * NEXP + t] * (1.0f / 4096.0f));
  red[t] = p;
  __syncthreads();
  for (int s = 128; s > 0; s >>= 1) {
    if (t < s) red[t] += red[t + s];
    __syncthreads();
  }
  if (t == 0) aux_out[0] = red[0] * 0.5f * 0.001f;
}

// ---------------------------------------------------------------------------
extern "C" void kernel_launch(void* const* d_in, const int* in_sizes, int n_in,
                              void* d_out, int out_size, void* d_ws, size_t ws_size,
                              hipStream_t stream) {
  const float* x    = (const float*)d_in[0];   // [2,4096,7168]
  const float* W    = (const float*)d_in[1];   // [256,7168]
  const float* bias = (const float*)d_in[2];   // [256]

  char* ws = (char*)d_ws;
  unsigned short* wsplit = (unsigned short*)ws;            // 224*32KB = 7,340,032 B
  float* logits = (float*)(ws + 7340032);                  // 8,388,608 B
  float* gsum   = (float*)(ws + 7340032 + 8388608);        // 2048 B
  float* gcnt   = gsum + 512;                              // 2048 B

  float* out  = (float*)d_out;
  float* wout = out;                 // [8192*8] routing weights
  float* iout = out + NTOK * 8;      // [8192*8] expert indices (as float)
  float* aux  = out + 2 * NTOK * 8;  // [1] aux loss

  (void)hipMemsetAsync(gsum, 0, 4096, stream);             // gsum+gcnt
  split_w_kernel<<<448, 256, 0, stream>>>(W, wsplit);
  gate_gemm_kernel<<<NTOK / 64, 256, 65536, stream>>>(x, wsplit, logits);
  topk_kernel<<<NTOK / 64, 256, 0, stream>>>(logits, bias, wout, iout, gsum, gcnt);
  finalize_kernel<<<1, 256, 0, stream>>>(gsum, gcnt, aux);
}
